// HSGBDHLevel_29171417874551
// MI455X (gfx1250) — matrix operand order repaired and checked by reading, not compiled, not run on hardware
//
#include <hip/hip_runtime.h>

// ---------------------------------------------------------------------------
// HSGBDH level kernel for MI455X (gfx1250, wave32, WMMA + TDM/async-LDS).
// GEMMs: v_wmma_f32_16x16x32_f16, fp32 accum, double-buffered LDS pipeline.
// ---------------------------------------------------------------------------

typedef __attribute__((ext_vector_type(16))) _Float16 v16h;
typedef __attribute__((ext_vector_type(8)))  _Float16 v8h;
typedef __attribute__((ext_vector_type(4)))  _Float16 v4h;
typedef __attribute__((ext_vector_type(8)))  float    v8f;
typedef __attribute__((ext_vector_type(4)))  unsigned int u32x4;
typedef __attribute__((ext_vector_type(8)))  int      i32x8;
typedef __attribute__((ext_vector_type(4)))  int      i32x4;

#define NK     4096
#define DDIM   1024
#define BATCH  8
#define EPSLN  1e-5f
#define THRESH 0.1f

#define LDA_S 40   // f16 elems per A row in LDS (80B, 16B-aligned, 32 data + 8 pad)
#define LDB_S 40   // f16 elems per B column in LDS

#define GLOBAL_AS __attribute__((address_space(1)))
#define LDS_AS    __attribute__((address_space(3)))

#if defined(__AMDGCN__) && __has_builtin(__builtin_amdgcn_tensor_load_to_lds)
#define HAS_TDM 1
#else
#define HAS_TDM 0
#endif
#if defined(__AMDGCN__) && __has_builtin(__builtin_amdgcn_global_load_async_to_lds_b128)
#define HAS_ASYNC 1
#else
#define HAS_ASYNC 0
#endif

__device__ inline void wait_asynccnt0() {
#if defined(__AMDGCN__)
#if __has_builtin(__builtin_amdgcn_s_wait_asynccnt)
  __builtin_amdgcn_s_wait_asynccnt(0);
#else
  asm volatile("s_wait_asynccnt 0x0" ::: "memory");
#endif
#endif
}

#if HAS_ASYNC
// global -> LDS async copy of 16 bytes (per active lane)
__device__ inline void async_copy_b128(const _Float16* g, _Float16* l) {
  __builtin_amdgcn_global_load_async_to_lds_b128(
      (GLOBAL_AS i32x4*)(size_t)g,
      (LDS_AS i32x4*)(unsigned int)(size_t)l, 0, 0);
}
#endif

#if HAS_TDM
// Issue one TDM 2D tile load: global (row-major, lda f16/row) -> LDS tile with
// 32 f16 per row + 8 f16 pad (pad_interval=16 DWORDs, pad_amount=4 DWORDs).
__device__ inline void tdm_load_tile_f16(const _Float16* gtile, unsigned int lds_off,
                                         int lda, int tile_rows) {
  unsigned long long ga = (unsigned long long)(size_t)gtile;
  u32x4 g0;
  g0[0] = 1u;                                             // count=1, user mode
  g0[1] = lds_off;                                        // LDS byte address
  g0[2] = (unsigned int)ga;                               // global_addr[31:0]
  g0[3] = (unsigned int)((ga >> 32) & 0x01FFFFFFu) | (2u << 30);  // addr[56:32] | type=2
  i32x8 g1;
  g1[0] = (1 << 16) | (1 << 20) | (3 << 22) | (3 << 25);  // data_size=2B, pad_en, 16DW interval, 4DW pad
  g1[1] = (int)(((unsigned)lda & 0xFFFFu) << 16);         // barrier_addr=0 | tensor_dim0 lo16
  g1[2] = (int)((unsigned)lda >> 16);                     // tensor_dim0 hi16 | tensor_dim1 lo16 (=0)
  g1[3] = (int)((1u << 20) >> 16) | (32 << 16);           // tensor_dim1(=1<<20) hi16 | tile_dim0=32
  g1[4] = tile_rows;                                      // tile_dim1 | tile_dim2=0
  g1[5] = lda;                                            // tensor_dim0_stride lo32
  g1[6] = 0;
  g1[7] = 0;
  i32x4 z = {0, 0, 0, 0};
#if defined(__clang_major__) && __clang_major__ >= 23
  i32x8 z8 = {0, 0, 0, 0, 0, 0, 0, 0};
  __builtin_amdgcn_tensor_load_to_lds(g0, g1, z, z, z8, 0);
#else
  __builtin_amdgcn_tensor_load_to_lds(g0, g1, z, z, 0);
#endif
}
#endif

// ---- fragment loaders (CDNA5 16-bit WMMA layouts), b128 LDS loads ---------
__device__ inline v16h load_a_frag(const _Float16* As, int m0, int lane) {
  const _Float16* row = As + (m0 + (lane & 15)) * LDA_S;
  const int half = lane >> 4;
  v8h lo = *(const v8h*)(row + half * 8);
  v8h hi = *(const v8h*)(row + 16 + half * 8);
  v16h a;
  #pragma unroll
  for (int e = 0; e < 8; ++e) { a[e] = lo[e]; a[e + 8] = hi[e]; }
  return a;
}
__device__ inline v16h load_b_frag(const _Float16* Bs, int n0, int lane) {
  const _Float16* col = Bs + (n0 + (lane & 15)) * LDB_S + (lane >> 4) * 8;
  v8h lo = *(const v8h*)(col);
  v8h hi = *(const v8h*)(col + 16);
  v16h b;
  #pragma unroll
  for (int e = 0; e < 8; ++e) { b[e] = lo[e]; b[e + 8] = hi[e]; }
  return b;
}

#define WMMA_F16(A, B, C) \
  __builtin_amdgcn_wmma_f32_16x16x32_f16(false, (A), false, (B), (short)0, (C), false, false)

// ---- synchronous A staging (fallback + float-convert paths) ---------------
template <int BM, int NT>
__device__ inline void stage_a_sync(_Float16* As, const _Float16* A, int lda,
                                    int row0, int k0, int tid) {
  #pragma unroll
  for (int it = 0; it < (BM * 4) / NT; ++it) {
    int idx = tid + it * NT;
    int c8 = (idx & 3) << 3, r = idx >> 2;
    *(v8h*)(As + r * LDA_S + c8) =
        *(const v8h*)(A + (size_t)(row0 + r) * lda + k0 + c8);
  }
}
template <int BM, int NT>
__device__ inline void stage_a_sync(_Float16* As, const float* A, int lda,
                                    int row0, int k0, int tid) {
  #pragma unroll
  for (int it = 0; it < (BM * 8) / NT; ++it) {
    int idx = tid + it * NT;
    int c4 = (idx & 7) << 2, r = idx >> 3;
    float4 v = *(const float4*)(A + (size_t)(row0 + r) * lda + k0 + c4);
    v4h h = { (_Float16)v.x, (_Float16)v.y, (_Float16)v.z, (_Float16)v.w };
    *(v4h*)(As + r * LDA_S + c4) = h;
  }
}

// ---- best-path A staging: TDM > async > sync (f16); sync-convert (f32) ----
template <int BM, int NT>
__device__ inline void stage_a_issue(_Float16* As, const _Float16* A, int lda,
                                     int row0, int k0, int tid, int wave) {
#if HAS_TDM
  if (wave == 0)
    tdm_load_tile_f16(A + (size_t)row0 * lda + k0,
                      (unsigned int)(size_t)As, lda, BM);
#elif HAS_ASYNC
  #pragma unroll
  for (int it = 0; it < (BM * 4) / NT; ++it) {
    int idx = tid + it * NT;
    int c8 = (idx & 3) << 3, r = idx >> 2;
    async_copy_b128(A + (size_t)(row0 + r) * lda + k0 + c8,
                    As + r * LDA_S + c8);
  }
#else
  stage_a_sync<BM, NT>(As, A, lda, row0, k0, tid);
#endif
  (void)wave; (void)tid;
}
template <int BM, int NT>
__device__ inline void stage_a_issue(_Float16* As, const float* A, int lda,
                                     int row0, int k0, int tid, int wave) {
  (void)wave;
  stage_a_sync<BM, NT>(As, A, lda, row0, k0, tid);
}
__device__ inline void stage_a_wait(const _Float16*, int wave) {
#if HAS_TDM
  if (wave == 0) __builtin_amdgcn_s_wait_tensorcnt(0);
#elif HAS_ASYNC
  wait_asynccnt0();
#endif
  (void)wave;
}
__device__ inline void stage_a_wait(const float*, int) {}

// ---- split B staging (NT=256): global [k][n] -> regs -> LDS [n][k] --------
// Each thread owns one 8-wide n-run at one k row: n8=(tid&7)*8, kk=tid>>3.
__device__ inline v8h b_load_global(const _Float16* B, int ldb, int k0, int col0,
                                    int tid) {
  int n8 = (tid & 7) << 3, kk = tid >> 3;
  return *(const v8h*)(B + (size_t)(k0 + kk) * ldb + col0 + n8);
}
__device__ inline v8h b_load_global(const float* B, int ldb, int k0, int col0,
                                    int tid) {
  int n8 = (tid & 7) << 3, kk = tid >> 3;
  const float* p = B + (size_t)(k0 + kk) * ldb + col0 + n8;
  float4 v0 = *(const float4*)(p);
  float4 v1 = *(const float4*)(p + 4);
  v8h g = { (_Float16)v0.x, (_Float16)v0.y, (_Float16)v0.z, (_Float16)v0.w,
            (_Float16)v1.x, (_Float16)v1.y, (_Float16)v1.z, (_Float16)v1.w };
  return g;
}
__device__ inline void b_store_lds(_Float16* Bs, v8h g, int tid) {
  int n8 = (tid & 7) << 3, kk = tid >> 3;
  #pragma unroll
  for (int e = 0; e < 8; ++e) Bs[(n8 + e) * LDB_S + kk] = g[e];
}

// ---------------------------------------------------------------------------
// 128x64-block WMMA GEMM, double-buffered: C = A[M,K] * B[K,N] (+ I).
// 256 threads = 8 waves (4x2); each wave computes 32x32 via 4 accumulators.
// ---------------------------------------------------------------------------
template <typename TA, typename TB>
__global__ __launch_bounds__(256)
void gemm128x64_wmma(const TA* __restrict__ A, int lda,
                     const TB* __restrict__ B, int ldb, size_t bstrideB,
                     _Float16* __restrict__ out16, float* __restrict__ out32,
                     int ldo, size_t bstrideO, int K, int addI)
{
  __shared__ __align__(16) _Float16 As[2][128 * LDA_S];
  __shared__ __align__(16) _Float16 Bs[2][64 * LDB_S];

  const int tid  = threadIdx.x;
  const int lane = tid & 31;
  const int wave = tid >> 5;
  const int wr = wave >> 1, wc = wave & 1;       // 4 x 2 waves
  const int row0 = blockIdx.y * 128;
  const int col0 = blockIdx.x * 64;
  B += (size_t)blockIdx.z * bstrideB;

  v8f acc0 = {}, acc1 = {}, acc2 = {}, acc3 = {};

  const int NIT = K / 32;
  // prologue: fully stage chunk 0
  stage_a_issue<128, 256>(As[0], A, lda, row0, 0, tid, wave);
  {
    v8h b0 = b_load_global(B, ldb, 0, col0, tid);
    b_store_lds(Bs[0], b0, tid);
  }

  for (int it = 0; it < NIT; ++it) {
    const int cur = it & 1;
    stage_a_wait(A, wave);        // A transfers for `cur` complete (per wave)
    __syncthreads();              // all staging for `cur` visible to all waves

    v8h bnext = {};
    const bool more = (it + 1) < NIT;
    if (more) {                   // issue next chunk; latency hidden by WMMAs
      stage_a_issue<128, 256>(As[cur ^ 1], A, lda, row0, (it + 1) * 32, tid, wave);
      bnext = b_load_global(B, ldb, (it + 1) * 32, col0, tid);
    }

    const _Float16* Ac = As[cur];
    const _Float16* Bc = Bs[cur];
    v16h a0 = load_a_frag(Ac, wr * 32,      lane);
    v16h a1 = load_a_frag(Ac, wr * 32 + 16, lane);
    v16h b0 = load_b_frag(Bc, wc * 32,      lane);
    v16h b1 = load_b_frag(Bc, wc * 32 + 16, lane);

    acc0 = WMMA_F16(a0, b0, acc0);
    acc1 = WMMA_F16(a0, b1, acc1);
    acc2 = WMMA_F16(a1, b0, acc2);
    acc3 = WMMA_F16(a1, b1, acc3);

    if (more) b_store_lds(Bs[cur ^ 1], bnext, tid);   // loadcnt wait lands here
  }

  if (out16) out16 += (size_t)blockIdx.z * bstrideO;
  const int half = lane >> 4;
  const int nlo  = lane & 15;

  #pragma unroll
  for (int t = 0; t < 4; ++t) {
    const int tr = t >> 1, tc = t & 1;
    v8f acc = (t == 0) ? acc0 : (t == 1) ? acc1 : (t == 2) ? acc2 : acc3;
    #pragma unroll
    for (int r = 0; r < 8; ++r) {
      int i = row0 + wr * 32 + tr * 16 + half * 8 + r;
      int j = col0 + wc * 32 + tc * 16 + nlo;
      float val = acc[r];
      if (addI && i == j) val += 1.0f;
      if (out32) out32[(size_t)i * ldo + j] = val;
      if (out16) out16[(size_t)i * ldo + j] = (_Float16)val;
    }
  }
}

// ---------------------------------------------------------------------------
// Fused gate + G construction (64x64 block, 128 threads, 2x2 waves).
// A tile staged via global_load_async_to_lds when available.
// ---------------------------------------------------------------------------
__global__ __launch_bounds__(128)
void gate_build_g(const _Float16* __restrict__ proj, const float* __restrict__ Dx,
                  const float* __restrict__ y0,
                  float* __restrict__ Gout, _Float16* __restrict__ G16)
{
  __shared__ __align__(16) _Float16 As[64 * LDA_S];
  __shared__ __align__(16) _Float16 Bs[64 * LDB_S];

  const int tid  = threadIdx.x;
  const int lane = tid & 31;
  const int wave = tid >> 5;
  const int wr = wave >> 1, wc = wave & 1;
  const int row0 = blockIdx.y * 64;
  const int col0 = blockIdx.x * 64;

  v8f gate0 = {}, gate1 = {}, gate2 = {}, gate3 = {};

  for (int h = 0; h < 3; ++h) {
    const _Float16* Ah = proj + (size_t)h * NK * DDIM;
    v8f acc0 = {}, acc1 = {}, acc2 = {}, acc3 = {};

    for (int k0 = 0; k0 < DDIM; k0 += 32) {
#if HAS_ASYNC
      #pragma unroll
      for (int it = 0; it < 2; ++it) {
        int idx = tid + it * 128;
        int c8 = (idx & 3) << 3, r = idx >> 2;
        async_copy_b128(Ah + (size_t)(row0 + r) * DDIM + k0 + c8,
                        As + r * LDA_S + c8);
      }
#else
      stage_a_sync<64, 128>(As, Ah, DDIM, row0, k0, tid);
#endif
      // B[k][n] = Dx[col0+n][k0+k]: float4 read along k, v4h store along k
      #pragma unroll
      for (int it = 0; it < 4; ++it) {
        int idx = tid + it * 128;
        int k4 = (idx & 7) << 2, n = idx >> 3;
        float4 v = *(const float4*)(Dx + (size_t)(col0 + n) * DDIM + k0 + k4);
        v4h hh = { (_Float16)v.x, (_Float16)v.y, (_Float16)v.z, (_Float16)v.w };
        *(v4h*)(Bs + n * LDB_S + k4) = hh;
      }
#if HAS_ASYNC
      wait_asynccnt0();
#endif
      __syncthreads();

      v16h a0 = load_a_frag(As, wr * 32,      lane);
      v16h a1 = load_a_frag(As, wr * 32 + 16, lane);
      v16h b0 = load_b_frag(Bs, wc * 32,      lane);
      v16h b1 = load_b_frag(Bs, wc * 32 + 16, lane);

      acc0 = WMMA_F16(a0, b0, acc0);
      acc1 = WMMA_F16(a0, b1, acc1);
      acc2 = WMMA_F16(a1, b0, acc2);
      acc3 = WMMA_F16(a1, b1, acc3);
      __syncthreads();
    }
    #pragma unroll
    for (int r = 0; r < 8; ++r) {   // fast sigmoid: v_exp + v_rcp
      gate0[r] += __builtin_amdgcn_rcpf(1.0f + __expf(-acc0[r]));
      gate1[r] += __builtin_amdgcn_rcpf(1.0f + __expf(-acc1[r]));
      gate2[r] += __builtin_amdgcn_rcpf(1.0f + __expf(-acc2[r]));
      gate3[r] += __builtin_amdgcn_rcpf(1.0f + __expf(-acc3[r]));
    }
  }

  const int half = lane >> 4;
  const int nlo  = lane & 15;
  #pragma unroll
  for (int t = 0; t < 4; ++t) {
    const int tr = t >> 1, tc = t & 1;
    v8f g = (t == 0) ? gate0 : (t == 1) ? gate1 : (t == 2) ? gate2 : gate3;
    #pragma unroll
    for (int r = 0; r < 8; ++r) {
      int i = row0 + wr * 32 + tr * 16 + half * 8 + r;
      int j = col0 + wc * 32 + tc * 16 + nlo;
      float gv = g[r] * (1.0f / 3.0f);
      float xv = y0[i];
      float yv = y0[j];
      float verified = xv * yv * gv;
      float outv = (xv > THRESH && yv > THRESH && i != j && verified > 0.0f)
                       ? verified : 0.0f;
      Gout[(size_t)i * NK + j] = outv;
      G16[(size_t)i * NK + j] = (_Float16)outv;
    }
  }
}

// ---------------------------------------------------------------------------
__global__ __launch_bounds__(256)
void encode_matvec(const float* __restrict__ x, const float* __restrict__ E,
                   float* __restrict__ V)
{
  int n = blockIdx.x * 256 + threadIdx.x;
  int b = blockIdx.y;
  const float* xb = x + b * DDIM;
  float s = 0.0f;
  for (int d = 0; d < DDIM; ++d) s += xb[d] * E[(size_t)d * NK + n];
  V[b * NK + n] = s;
}

__global__ __launch_bounds__(256)
void layernorm_relu(const float* __restrict__ V, float* __restrict__ Y)
{
  __shared__ float s1[256];
  __shared__ float s2[256];
  int b = blockIdx.x, tid = threadIdx.x;
  const float* v = V + (size_t)b * NK;
  float a = 0.0f, q = 0.0f;
  for (int n = tid; n < NK; n += 256) { float t = v[n]; a += t; q += t * t; }
  s1[tid] = a; s2[tid] = q;
  __syncthreads();
  for (int off = 128; off > 0; off >>= 1) {
    if (tid < off) { s1[tid] += s1[tid + off]; s2[tid] += s2[tid + off]; }
    __syncthreads();
  }
  float mu  = s1[0] * (1.0f / NK);
  float var = s2[0] * (1.0f / NK) - mu * mu;
  float inv = rsqrtf(var + EPSLN);
  for (int n = tid; n < NK; n += 256) {
    float t = (v[n] - mu) * inv;
    Y[(size_t)b * NK + n] = t > 0.0f ? t : 0.0f;
  }
}

__global__ __launch_bounds__(256)
void init_r1(const _Float16* __restrict__ G16, _Float16* __restrict__ R)
{
  size_t idx = (size_t)blockIdx.x * 256 + threadIdx.x;
  int i = (int)(idx >> 12);
  int j = (int)(idx & (NK - 1));
  float g = (float)G16[idx];
  R[idx] = (_Float16)(g + (i == j ? 1.0f : 0.0f));
}

// ---------------------------------------------------------------------------
extern "C" void kernel_launch(void* const* d_in, const int* in_sizes, int n_in,
                              void* d_out, int out_size, void* d_ws, size_t ws_size,
                              hipStream_t stream)
{
  (void)in_sizes; (void)n_in; (void)out_size; (void)ws_size;
  const float* x_t = (const float*)d_in[0];   // (8, 1024)
  const float* E   = (const float*)d_in[1];   // (1024, 4096)
  const float* Dx  = (const float*)d_in[2];   // (4096, 1024)
  const float* W   = (const float*)d_in[3];   // (3, 1024, 1024)

  float* y_out = (float*)d_out;                      // (8, 4096)
  float* G_out = y_out + (size_t)BATCH * NK;         // (4096, 4096)
  float* R_out = G_out + (size_t)NK * NK;            // (4096, 4096)

  char* w = (char*)d_ws;
  size_t off = 0;
  float*     ws_v    = (float*)(w + off);     off += (size_t)BATCH * NK * sizeof(float);
  _Float16*  ws_proj = (_Float16*)(w + off);  off += (size_t)3 * NK * DDIM * sizeof(_Float16);
  _Float16*  ws_G    = (_Float16*)(w + off);  off += (size_t)NK * NK * sizeof(_Float16);
  _Float16*  ws_RA   = (_Float16*)(w + off);  off += (size_t)NK * NK * sizeof(_Float16);
  _Float16*  ws_RB   = (_Float16*)(w + off);

  // 1) encode + layernorm + relu -> y_t
  encode_matvec<<<dim3(NK / 256, BATCH), 256, 0, stream>>>(x_t, E, ws_v);
  layernorm_relu<<<BATCH, 256, 0, stream>>>(ws_v, y_out);

  // 2) proj[h] = Dx @ W[h]  (batched over blockIdx.z), f16 output
  gemm128x64_wmma<float, float><<<dim3(DDIM / 64, NK / 128, 3), 256, 0, stream>>>(
      Dx, DDIM, W, DDIM, (size_t)DDIM * DDIM,
      ws_proj, nullptr, DDIM, (size_t)NK * DDIM, DDIM, 0);

  // 3) gate = mean_h sigmoid(proj[h] @ Dx^T); fuse mask -> G (fp32 + f16)
  gate_build_g<<<dim3(NK / 64, NK / 64), 128, 0, stream>>>(
      ws_proj, Dx, y_out, G_out, ws_G);

  // 4) closure: R1 = I + G; then R_{k+1} = I + G @ R_k (Horner), 5 total
  init_r1<<<(NK * NK) / 256, 256, 0, stream>>>(ws_G, ws_RA);
  gemm128x64_wmma<_Float16, _Float16><<<dim3(NK / 64, NK / 128), 256, 0, stream>>>(
      ws_G, NK, ws_RA, NK, 0, ws_RB, nullptr, NK, 0, NK, 1);          // R2
  gemm128x64_wmma<_Float16, _Float16><<<dim3(NK / 64, NK / 128), 256, 0, stream>>>(
      ws_G, NK, ws_RB, NK, 0, ws_RA, nullptr, NK, 0, NK, 1);          // R3
  gemm128x64_wmma<_Float16, _Float16><<<dim3(NK / 64, NK / 128), 256, 0, stream>>>(
      ws_G, NK, ws_RA, NK, 0, ws_RB, nullptr, NK, 0, NK, 1);          // R4
  gemm128x64_wmma<_Float16, _Float16><<<dim3(NK / 64, NK / 128), 256, 0, stream>>>(
      ws_G, NK, ws_RB, NK, 0, nullptr, R_out, NK, 0, NK, 1);          // R5 -> out
}
